// DASEN_v1_2860448220006
// MI455X (gfx1250) — compile-verified, matrix operand order checked by reading, not run on hardware
//
#include <hip/hip_runtime.h>
#include <hip/hip_bf16.h>

// ---------------- constants from the reference ----------------
#define BATCH     32
#define NTOK      510
#define TSEQ      512          // 510 tokens + hidden + skill
#define EMB       64
#define HEADS     8
#define DEPTH     2
#define INNER     (EMB*HEADS)  // 512
#define FFDIM     (4*EMB)      // 256
#define SKILLN    16
#define SKILLH    128
#define NENEMY    255
#define MROWS     (BATCH*TSEQ) // 16384
#define KCHUNK    64           // K-chunk staged in LDS per iteration

typedef __attribute__((ext_vector_type(16))) _Float16 v16h;
typedef __attribute__((ext_vector_type(8)))  _Float16 v8h;
typedef __attribute__((ext_vector_type(8)))  float    v8f;

union AccU { v8f v; float f[8]; };

// ---------------- WMMA helpers ----------------
// A-fragment / B-fragment (B stored transposed as [N][K] row-major) load.
// Per CDNA5 ISA 16-bit A layout (16x32, MxK): lane l(0..15) row M=l,
// halves 0..7  = K[k0 + hi*8 .. +8), halves 8..15 = K[k0+16+hi*8 .. +8)
// where hi = lane>>4. Two contiguous 16B chunks per lane.
__device__ __forceinline__ v16h load_frag(const _Float16* p, int ld,
                                          int row0, int k0, int lane) {
  int l  = lane & 15;
  int hi = (lane >> 4) & 1;
  const _Float16* r = p + (size_t)(row0 + l) * ld + k0 + hi * 8;
  union { v16h v; v8h h[2]; } u;
  u.h[0] = *reinterpret_cast<const v8h*>(r);
  u.h[1] = *reinterpret_cast<const v8h*>(r + 16);
  return u.v;
}

__device__ __forceinline__ v8f wmma32(v16h a, v16h b, v8f c) {
  // 8 args: (neg_a, A, neg_b, B, c_mod, C, reuse_a, reuse_b)
  return __builtin_amdgcn_wmma_f32_16x16x32_f16(false, a, false, b,
                                                (short)0, c, false, false);
}

// gfx1250 async global->LDS copy, 16B per lane, tracked by ASYNCcnt.
__device__ __forceinline__ void async_copy_b128(unsigned lds_byte_addr,
                                                const _Float16* gptr) {
  asm volatile("global_load_async_to_lds_b128 %0, %1, off"
               :: "v"(lds_byte_addr), "v"(gptr) : "memory");
}
__device__ __forceinline__ void wait_asynccnt0() {
  asm volatile("s_wait_asynccnt 0" ::: "memory");
}

// ---------------- weight prep ----------------
__global__ void k_cvt(const float* __restrict__ in, _Float16* __restrict__ out, int n) {
  int i = blockIdx.x * blockDim.x + threadIdx.x;
  if (i < n) out[i] = (_Float16)in[i];
}

// in: [D][K][N] fp32 row-major -> out: [D][N][K] f16 (transposed per slab)
__global__ void k_cvt_tr(const float* __restrict__ in, _Float16* __restrict__ out,
                         int K, int N, int total) {
  int i = blockIdx.x * blockDim.x + threadIdx.x;
  if (i >= total) return;
  int nk = N * K;
  int d = i / nk, r = i % nk;
  int n = r / K, k = r % K;
  out[i] = (_Float16)in[(size_t)d * nk + (size_t)k * N + n];
}

// ---------------- LDS-staged WMMA GEMM ----------------
// out[M][N] = A[M][K](f16) * Bt[N][K](f16)^T (+bias)(+relu)(+resid)
// Block = 8 waves = 128 output rows x one 64-col panel.
// Weight panel (64 x KCHUNK halves) staged via global_load_async_to_lds_b128;
// each wave computes 4 x (16x16) tiles, reusing its A fragment 4x.
__global__ void k_gemm_lds(const _Float16* __restrict__ A, int lda,
                           const _Float16* __restrict__ Bt, int ldb,
                           const float* __restrict__ bias,
                           const float* __restrict__ resid,
                           float* __restrict__ outF,
                           _Float16* __restrict__ outH,
                           int M, int N, int K, int relu, int embed_remap) {
  __shared__ __align__(16) _Float16 Bl[64 * KCHUNK];   // 8 KB weight panel
  int panels = N >> 6;
  int panel  = blockIdx.x % panels;
  int rowblk = blockIdx.x / panels;
  int wv   = threadIdx.x >> 5;
  int lane = threadIdx.x & 31;
  int tid  = threadIdx.x;
  int row0 = rowblk * 128 + wv * 16;
  bool active = (row0 < M);                 // wave-uniform (M % 16 == 0)
  int colbase = panel << 6;
  unsigned ldsBase = (unsigned)(uintptr_t)(void*)&Bl[0];

  AccU c[4];
#pragma unroll
  for (int j = 0; j < 4; ++j)
#pragma unroll
    for (int r = 0; r < 8; ++r) c[j].f[r] = 0.0f;

  for (int kc = 0; kc < K; kc += KCHUNK) {
    if (kc) __syncthreads();                // previous chunk fully consumed
    // stage 64 rows x KCHUNK halves = 512 x 16B : 2 async copies per thread
#pragma unroll
    for (int i = 0; i < 2; ++i) {
      int cid  = tid + i * 256;             // 0..511
      int rrow = cid >> 3;
      int rk   = (cid & 7) * 8;
      const _Float16* src = Bt + (size_t)(colbase + rrow) * ldb + kc + rk;
      async_copy_b128(ldsBase + (unsigned)((rrow * KCHUNK + rk) * 2), src);
    }
    wait_asynccnt0();                       // my async writes landed in LDS
    __syncthreads();                        // panel visible to all waves
    if (active) {
#pragma unroll
      for (int ks = 0; ks < KCHUNK; ks += 32) {
        v16h a = load_frag(A, lda, row0, kc + ks, lane);
#pragma unroll
        for (int j = 0; j < 4; ++j) {
          v16h b = load_frag(Bl, KCHUNK, j * 16, ks, lane);
          c[j].v = wmma32(a, b, c[j].v);
        }
      }
    }
  }
  if (!active) return;

  int n = lane & 15, hi = (lane >> 4) & 1;
#pragma unroll
  for (int j = 0; j < 4; ++j) {
    int colg = colbase + j * 16 + n;
    float bv = bias ? bias[colg] : 0.0f;
#pragma unroll
    for (int r = 0; r < 8; ++r) {
      int row = row0 + hi * 8 + r;
      float v = c[j].f[r] + bv;
      if (relu) v = fmaxf(v, 0.0f);
      int orow = row;
      if (embed_remap) orow = (row / NTOK) * TSEQ + (row % NTOK);
      size_t oi = (size_t)orow * N + colg;
      if (resid) v += resid[oi];
      if (outF) outF[oi] = v;
      if (outH) outH[oi] = (_Float16)v;
    }
  }
}

// V projection with transposed store: vT[b][h][dim][key]
__global__ void k_gemm_vT(const _Float16* __restrict__ A,
                          const _Float16* __restrict__ Bt,
                          _Float16* __restrict__ vT) {
  int lane = threadIdx.x & 31;
  int wave = blockIdx.x * (blockDim.x >> 5) + (threadIdx.x >> 5);
  const int tilesN = INNER >> 4;                 // 32
  const int tiles  = (MROWS >> 4) * tilesN;      // 32768
  if (wave >= tiles) return;
  int row0 = (wave / tilesN) << 4;
  int col0 = (wave % tilesN) << 4;
  AccU c;
#pragma unroll
  for (int r = 0; r < 8; ++r) c.f[r] = 0.0f;
  for (int k0 = 0; k0 < EMB; k0 += 32) {
    v16h a = load_frag(A,  EMB, row0, k0, lane);
    v16h b = load_frag(Bt, EMB, col0, k0, lane);
    c.v = wmma32(a, b, c.v);
  }
  int n = lane & 15, hi = (lane >> 4) & 1;
  int ng  = col0 + n;              // 0..511 -> (h, dim)
  int h   = ng >> 6, dim = ng & 63;
  int b   = row0 >> 9;             // tile never crosses a batch (512-row) boundary
  int tok0 = (row0 & 511) + hi * 8;
  _Float16* dst = vT + (((size_t)(b * HEADS + h) * EMB + dim) * TSEQ) + tok0;
#pragma unroll
  for (int r = 0; r < 8; ++r) dst[r] = (_Float16)c.f[r];
}

// ---------------- embedding tail rows (hidden + skill embed) ----------------
__global__ void k_embed_tail(const float* __restrict__ hidden,
                             const float* __restrict__ skill_state,
                             const float* __restrict__ W_sdec,
                             const float* __restrict__ b_sdec,
                             float* __restrict__ x32, _Float16* __restrict__ xh) {
  int i = blockIdx.x * blockDim.x + threadIdx.x;
  if (i >= BATCH * 2 * EMB) return;
  int n = i & 63, t = (i >> 6) & 1, b = i >> 7;
  float v;
  if (t == 0) v = hidden[b * EMB + n];
  else {
    v = b_sdec[n];
    for (int k = 0; k < SKILLN; ++k) v += skill_state[b * SKILLN + k] * W_sdec[k * EMB + n];
  }
  size_t oi = ((size_t)(b * TSEQ + NTOK + t)) * EMB + n;
  x32[oi] = v;  xh[oi] = (_Float16)v;
}

// ---------------- flash attention (one wave = b, head, 16 q-rows) ----------------
__global__ void k_attn(const _Float16* __restrict__ q,
                       const _Float16* __restrict__ kk,
                       const _Float16* __restrict__ vT,
                       _Float16* __restrict__ attout) {
  __shared__ __align__(16) _Float16 Plds[8][16 * 32];   // per-wave P tile
  int lane = threadIdx.x & 31;
  int wave = blockIdx.x * (blockDim.x >> 5) + (threadIdx.x >> 5);
  const int QT = TSEQ / 16;                 // 32 query tiles
  const int total = BATCH * HEADS * QT;     // 8192 waves
  if (wave >= total) return;
  int qt = wave % QT;
  int bh = wave / QT;
  int h  = bh % HEADS, b = bh / HEADS;
  int qrow0 = b * TSEQ + qt * 16;
  _Float16* P = &Plds[threadIdx.x >> 5][0];

  v16h qa0 = load_frag(q, INNER, qrow0, h * EMB +  0, lane);
  v16h qa1 = load_frag(q, INNER, qrow0, h * EMB + 32, lane);

  AccU o0, o1, o2, o3;
  float mrow[8], lrow[8];
#pragma unroll
  for (int r = 0; r < 8; ++r) {
    o0.f[r] = o1.f[r] = o2.f[r] = o3.f[r] = 0.0f;
    mrow[r] = -3.0e38f;  lrow[r] = 0.0f;
  }

  const float sc = 0.125f;                  // 1/sqrt(64)
  for (int kt = 0; kt < TSEQ; kt += 32) {
    int krow0 = b * TSEQ + kt;
    if (kt + 32 < TSEQ)
      __builtin_prefetch(kk + (size_t)(krow0 + 32) * INNER + h * EMB, 0, 1);

    AccU s0, s1;
#pragma unroll
    for (int r = 0; r < 8; ++r) { s0.f[r] = 0.0f; s1.f[r] = 0.0f; }
    s0.v = wmma32(qa0, load_frag(kk, INNER, krow0,      h * EMB +  0, lane), s0.v);
    s0.v = wmma32(qa1, load_frag(kk, INNER, krow0,      h * EMB + 32, lane), s0.v);
    s1.v = wmma32(qa0, load_frag(kk, INNER, krow0 + 16, h * EMB +  0, lane), s1.v);
    s1.v = wmma32(qa1, load_frag(kk, INNER, krow0 + 16, h * EMB + 32, lane), s1.v);

    // online softmax per row (row r lives across 16 lanes of this half-wave)
#pragma unroll
    for (int r = 0; r < 8; ++r) {
      float a = s0.f[r] * sc, bb = s1.f[r] * sc;
      float mx = fmaxf(a, bb);
      for (int off = 1; off < 16; off <<= 1) mx = fmaxf(mx, __shfl_xor(mx, off, 16));
      float mn = fmaxf(mrow[r], mx);
      float c0 = __expf(mrow[r] - mn);
      mrow[r] = mn;
      float p0 = __expf(a - mn), p1 = __expf(bb - mn);
      float rs = p0 + p1;
      for (int off = 1; off < 16; off <<= 1) rs += __shfl_xor(rs, off, 16);
      lrow[r] = lrow[r] * c0 + rs;
      s0.f[r] = p0;  s1.f[r] = p1;
      o0.f[r] *= c0; o1.f[r] *= c0; o2.f[r] *= c0; o3.f[r] *= c0;
    }

    // C-layout -> A-layout relayout of P through per-wave LDS
    {
      int n = lane & 15, hi = (lane >> 4) & 1;
#pragma unroll
      for (int r = 0; r < 8; ++r) {
        P[(hi * 8 + r) * 32 + n]      = (_Float16)s0.f[r];
        P[(hi * 8 + r) * 32 + 16 + n] = (_Float16)s1.f[r];
      }
    }
    v16h pa = load_frag(P, 32, 0, 0, lane);

    int vbase = (b * HEADS + h) * EMB;
    o0.v = wmma32(pa, load_frag(vT, TSEQ, vbase +  0, kt, lane), o0.v);
    o1.v = wmma32(pa, load_frag(vT, TSEQ, vbase + 16, kt, lane), o1.v);
    o2.v = wmma32(pa, load_frag(vT, TSEQ, vbase + 32, kt, lane), o2.v);
    o3.v = wmma32(pa, load_frag(vT, TSEQ, vbase + 48, kt, lane), o3.v);
  }

  int n = lane & 15, hi = (lane >> 4) & 1;
#pragma unroll
  for (int r = 0; r < 8; ++r) {
    float inv = 1.0f / lrow[r];
    size_t row = (size_t)(qrow0 + hi * 8 + r) * INNER + h * EMB + n;
    attout[row +  0] = (_Float16)(o0.f[r] * inv);
    attout[row + 16] = (_Float16)(o1.f[r] * inv);
    attout[row + 32] = (_Float16)(o2.f[r] * inv);
    attout[row + 48] = (_Float16)(o3.f[r] * inv);
  }
}

// ---------------- layernorm (one wave per 64-wide row) ----------------
__global__ void k_layernorm(const float* __restrict__ y,
                            const float* __restrict__ g,
                            const float* __restrict__ be,
                            float* __restrict__ x32, _Float16* __restrict__ xh) {
  int wave = blockIdx.x * (blockDim.x >> 5) + (threadIdx.x >> 5);
  if (wave >= MROWS) return;
  int lane = threadIdx.x & 31;
  const float* row = y + (size_t)wave * EMB;
  float a = row[lane], b = row[lane + 32];
  float s = a + b;
  for (int off = 1; off < 32; off <<= 1) s += __shfl_xor(s, off, 32);
  float mean = s * (1.0f / EMB);
  float da = a - mean, db = b - mean;
  float v = da * da + db * db;
  for (int off = 1; off < 32; off <<= 1) v += __shfl_xor(v, off, 32);
  float inv = rsqrtf(v * (1.0f / EMB) + 1e-5f);
  float o0 = da * inv * g[lane]      + be[lane];
  float o1 = db * inv * g[lane + 32] + be[lane + 32];
  size_t oi = (size_t)wave * EMB;
  x32[oi + lane]      = o0;  xh[oi + lane]      = (_Float16)o0;
  x32[oi + lane + 32] = o1;  xh[oi + lane + 32] = (_Float16)o1;
}

// ---------------- output heads ----------------
__global__ void k_heads(const float* __restrict__ out32,
                        const float* __restrict__ W_qself, const float* __restrict__ b_qself,
                        const float* __restrict__ W_fc1,   const float* __restrict__ b_fc1,
                        const float* __restrict__ W_fc2,   const float* __restrict__ b_fc2,
                        const float* __restrict__ gumbel,  const int* __restrict__ enemy_num_p,
                        float* __restrict__ qout, float* __restrict__ hout,
                        float* __restrict__ skillout) {
  int b = blockIdx.x, t = threadIdx.x;      // 64 threads
  int nEnemy = enemy_num_p[0];
  __shared__ float s[EMB];
  __shared__ float sp[6];
  __shared__ float wsum[EMB];
  __shared__ float hid[SKILLH];
  __shared__ float spsum_sh;
  const float* rowbase = out32 + (size_t)b * TSEQ * EMB;
  s[t] = rowbase[(size_t)(TSEQ - 1) * EMB + t];
  hout[b * EMB + t] = rowbase[(size_t)(TSEQ - 2) * EMB + t];
  float ws = 0.0f;
  for (int j = 0; j < 6; ++j) ws += W_qself[t * 6 + j];
  wsum[t] = ws;
  __syncthreads();
  if (t < 6) {
    float acc = b_qself[t];
    for (int n = 0; n < EMB; ++n) acc += s[n] * W_qself[(EMB + n) * 6 + t];
    sp[t] = acc;
    float qb = acc;
    for (int n = 0; n < EMB; ++n) qb += rowbase[n] * W_qself[n * 6 + t];
    qout[b * (6 + NENEMY) + t] = qb;
  }
  __syncthreads();
  if (t == 0) { float a = 0; for (int j = 0; j < 6; ++j) a += sp[j]; spsum_sh = a; }
  __syncthreads();
  for (int e = t; e < nEnemy; e += 64) {
    const float* er = rowbase + (size_t)(1 + e) * EMB;
    float acc = 0.0f;
    for (int n = 0; n < EMB; ++n) acc += er[n] * wsum[n];
    qout[b * (6 + NENEMY) + 6 + e] = (acc + spsum_sh) * (1.0f / 6.0f);
  }
  for (int hh = t; hh < SKILLH; hh += 64) {
    float acc = b_fc1[hh];
    for (int n = 0; n < EMB; ++n) acc += s[n] * W_fc1[n * SKILLH + hh];
    hid[hh] = fmaxf(acc, 0.0f);
  }
  __syncthreads();
  if (t == 0) {
    float logit[SKILLN];
    float mx = -3.0e38f;
    for (int k = 0; k < SKILLN; ++k) {
      float acc = b_fc2[k];
      for (int hh = 0; hh < SKILLH; ++hh) acc += hid[hh] * W_fc2[hh * SKILLN + k];
      logit[k] = acc + gumbel[b * SKILLN + k];
      mx = fmaxf(mx, logit[k]);
    }
    float sum = 0.0f, ex[SKILLN];
    for (int k = 0; k < SKILLN; ++k) { ex[k] = __expf(logit[k] - mx); sum += ex[k]; }
    float inv = 1.0f / sum;
    for (int k = 0; k < SKILLN; ++k) skillout[b * SKILLN + k] = ex[k] * inv;
  }
}

// ---------------- host launch ----------------
static inline int blk(int work, int per) { return (work + per - 1) / per; }

extern "C" void kernel_launch(void* const* d_in, const int* in_sizes, int n_in,
                              void* d_out, int out_size, void* d_ws, size_t ws_size,
                              hipStream_t stream) {
  const float* inputs     = (const float*)d_in[0];
  const float* hidden     = (const float*)d_in[1];
  const float* skillst    = (const float*)d_in[2];
  const float* gumbel     = (const float*)d_in[3];
  const float* W_tok      = (const float*)d_in[4];
  const float* b_tok      = (const float*)d_in[5];
  const float* W_sdec     = (const float*)d_in[6];
  const float* b_sdec     = (const float*)d_in[7];
  const float* Wq         = (const float*)d_in[8];
  const float* Wk         = (const float*)d_in[9];
  const float* Wv         = (const float*)d_in[10];
  const float* Wu         = (const float*)d_in[11];
  const float* bu         = (const float*)d_in[12];
  const float* g1         = (const float*)d_in[13];
  const float* be1        = (const float*)d_in[14];
  const float* g2         = (const float*)d_in[15];
  const float* be2        = (const float*)d_in[16];
  const float* W_ff1      = (const float*)d_in[17];
  const float* b_ff1      = (const float*)d_in[18];
  const float* W_ff2      = (const float*)d_in[19];
  const float* b_ff2      = (const float*)d_in[20];
  const float* W_top      = (const float*)d_in[21];
  const float* b_top      = (const float*)d_in[22];
  const float* W_qself    = (const float*)d_in[23];
  const float* b_qself    = (const float*)d_in[24];
  const float* W_fc1      = (const float*)d_in[25];
  const float* b_fc1      = (const float*)d_in[26];
  const float* W_fc2      = (const float*)d_in[27];
  const float* b_fc2      = (const float*)d_in[28];
  const int*   enemy_num  = (const int*)d_in[29];

  // workspace carving (256B aligned)
  char* base = (char*)d_ws;
  size_t off = 0;
  auto carve = [&](size_t bytes) -> char* {
    off = (off + 255) & ~(size_t)255;
    char* p = base + off;
    off += bytes;
    return p;
  };
  _Float16* inp_h  = (_Float16*)carve((size_t)BATCH * NTOK * EMB * 2);
  _Float16* wtok_t = (_Float16*)carve(EMB * EMB * 2);
  _Float16* wq_t   = (_Float16*)carve((size_t)DEPTH * INNER * EMB * 2);
  _Float16* wk_t   = (_Float16*)carve((size_t)DEPTH * INNER * EMB * 2);
  _Float16* wv_t   = (_Float16*)carve((size_t)DEPTH * INNER * EMB * 2);
  _Float16* wu_t   = (_Float16*)carve((size_t)DEPTH * EMB * INNER * 2);
  _Float16* wff1_t = (_Float16*)carve((size_t)DEPTH * FFDIM * EMB * 2);
  _Float16* wff2_t = (_Float16*)carve((size_t)DEPTH * EMB * FFDIM * 2);
  _Float16* wtop_t = (_Float16*)carve(EMB * EMB * 2);
  float*    x32    = (float*)   carve((size_t)MROWS * EMB * 4);
  _Float16* xh     = (_Float16*)carve((size_t)MROWS * EMB * 2);
  float*    y32    = (float*)   carve((size_t)MROWS * EMB * 4);
  _Float16* qb     = (_Float16*)carve((size_t)MROWS * INNER * 2);
  _Float16* kb     = (_Float16*)carve((size_t)MROWS * INNER * 2);
  _Float16* vT     = (_Float16*)carve((size_t)MROWS * INNER * 2);
  _Float16* attout = (_Float16*)carve((size_t)MROWS * INNER * 2);
  _Float16* ffh    = (_Float16*)carve((size_t)MROWS * FFDIM * 2);
  float*    out32  = (float*)   carve((size_t)MROWS * EMB * 4);
  (void)ws_size; (void)in_sizes; (void)n_in; (void)out_size;

  float* qout     = (float*)d_out;                         // 32*261
  float* hout     = qout + BATCH * (6 + NENEMY);           // 32*64
  float* skillout = hout + BATCH * EMB;                    // 32*16

  const int TPB = 256;

  // --- weight / input conversion ---
  { int n = BATCH * NTOK * EMB;
    k_cvt<<<blk(n, TPB), TPB, 0, stream>>>(inputs, inp_h, n); }
  k_cvt_tr<<<blk(EMB*EMB, TPB), TPB, 0, stream>>>(W_tok, wtok_t, EMB, EMB, EMB*EMB);
  k_cvt_tr<<<blk(DEPTH*INNER*EMB, TPB), TPB, 0, stream>>>(Wq, wq_t, EMB, INNER, DEPTH*INNER*EMB);
  k_cvt_tr<<<blk(DEPTH*INNER*EMB, TPB), TPB, 0, stream>>>(Wk, wk_t, EMB, INNER, DEPTH*INNER*EMB);
  k_cvt_tr<<<blk(DEPTH*INNER*EMB, TPB), TPB, 0, stream>>>(Wv, wv_t, EMB, INNER, DEPTH*INNER*EMB);
  k_cvt_tr<<<blk(DEPTH*EMB*INNER, TPB), TPB, 0, stream>>>(Wu, wu_t, INNER, EMB, DEPTH*EMB*INNER);
  k_cvt_tr<<<blk(DEPTH*FFDIM*EMB, TPB), TPB, 0, stream>>>(W_ff1, wff1_t, EMB, FFDIM, DEPTH*FFDIM*EMB);
  k_cvt_tr<<<blk(DEPTH*EMB*FFDIM, TPB), TPB, 0, stream>>>(W_ff2, wff2_t, FFDIM, EMB, DEPTH*EMB*FFDIM);
  k_cvt_tr<<<blk(EMB*EMB, TPB), TPB, 0, stream>>>(W_top, wtop_t, EMB, EMB, EMB*EMB);

  // --- embedding: tokens = inputs @ W_tok + b_tok (row-remapped 510 -> 512) ---
  { int grid = blk(BATCH * NTOK, 128) * (EMB / 64);     // 128 blocks
    k_gemm_lds<<<grid, TPB, 0, stream>>>(inp_h, EMB, wtok_t, EMB, b_tok, nullptr,
                                         x32, xh, BATCH * NTOK, EMB, EMB, 0, 1); }
  k_embed_tail<<<blk(BATCH*2*EMB, TPB), TPB, 0, stream>>>(hidden, skillst, W_sdec, b_sdec, x32, xh);

  // --- transformer layers ---
  for (int i = 0; i < DEPTH; ++i) {
    const _Float16* wqi = wq_t + (size_t)i * INNER * EMB;
    const _Float16* wki = wk_t + (size_t)i * INNER * EMB;
    const _Float16* wvi = wv_t + (size_t)i * INNER * EMB;
    const _Float16* wui = wu_t + (size_t)i * EMB * INNER;
    const _Float16* w1i = wff1_t + (size_t)i * FFDIM * EMB;
    const _Float16* w2i = wff2_t + (size_t)i * EMB * FFDIM;

    int gQKV = (MROWS / 128) * (INNER / 64);             // 1024 blocks
    k_gemm_lds<<<gQKV, TPB, 0, stream>>>(xh, EMB, wqi, EMB, nullptr, nullptr,
                                         nullptr, qb, MROWS, INNER, EMB, 0, 0);
    k_gemm_lds<<<gQKV, TPB, 0, stream>>>(xh, EMB, wki, EMB, nullptr, nullptr,
                                         nullptr, kb, MROWS, INNER, EMB, 0, 0);
    { int tV = (MROWS/16) * (INNER/16);                  // 32768 waves
      k_gemm_vT<<<blk(tV, 8), TPB, 0, stream>>>(xh, wvi, vT); }

    int tATT = BATCH * HEADS * (TSEQ/16);                // 8192 waves
    k_attn<<<blk(tATT, 8), TPB, 0, stream>>>(qb, kb, vT, attout);

    int gPROJ = (MROWS / 128) * (EMB / 64);              // 128 blocks
    k_gemm_lds<<<gPROJ, TPB, 0, stream>>>(attout, INNER, wui, INNER, bu + i*EMB, x32,
                                          y32, nullptr, MROWS, EMB, INNER, 0, 0);
    k_layernorm<<<blk(MROWS, 8), TPB, 0, stream>>>(y32, g1 + i*EMB, be1 + i*EMB, x32, xh);

    int gFF1 = (MROWS / 128) * (FFDIM / 64);             // 512 blocks
    k_gemm_lds<<<gFF1, TPB, 0, stream>>>(xh, EMB, w1i, EMB, b_ff1 + i*FFDIM, nullptr,
                                         nullptr, ffh, MROWS, FFDIM, EMB, 1, 0);
    k_gemm_lds<<<gPROJ, TPB, 0, stream>>>(ffh, FFDIM, w2i, FFDIM, b_ff2 + i*EMB, x32,
                                          y32, nullptr, MROWS, EMB, FFDIM, 0, 0);
    k_layernorm<<<blk(MROWS, 8), TPB, 0, stream>>>(y32, g2 + i*EMB, be2 + i*EMB, x32, xh);
  }

  // --- top projection + heads ---
  { int gTOP = (MROWS / 128) * (EMB / 64);               // 128 blocks
    k_gemm_lds<<<gTOP, TPB, 0, stream>>>(xh, EMB, wtop_t, EMB, b_top, nullptr,
                                         out32, nullptr, MROWS, EMB, EMB, 0, 0); }
  k_heads<<<BATCH, 64, 0, stream>>>(out32, W_qself, b_qself, W_fc1, b_fc1,
                                    W_fc2, b_fc2, gumbel, enemy_num,
                                    qout, hout, skillout);
}
